// LcRnnCell_3934190043339
// MI455X (gfx1250) — matrix operand order repaired and checked by reading, not compiled, not run on hardware
//
#include <hip/hip_runtime.h>
#include <hip/hip_bf16.h>

typedef __bf16 bf16_t;
typedef __attribute__((ext_vector_type(16))) __bf16 v16bf;
typedef __attribute__((ext_vector_type(8)))  float  v8f;

#define NB 4096      // batch rows
#define NH 1024      // hidden

__device__ __forceinline__ float gatef(float pd, float cond, float branch, float carry) {
    float eq = (pd == cond) ? 1.0f : 0.0f;
    float gt = (pd >  cond) ? 1.0f : 0.0f;
    return eq * branch + gt * carry;
}

// CDNA5 async copy: global -> LDS, 16B per lane, tracked by ASYNCcnt.
// lds_off: wave-relative LDS byte address (low 32 bits of generic shared ptr).
__device__ __forceinline__ void async_ld16(unsigned lds_off, const void* gptr) {
    asm volatile("global_load_async_to_lds_b128 %0, %1, off"
                 :: "v"(lds_off), "v"((unsigned long long)(uintptr_t)gptr)
                 : "memory");
}

union PackBF { bf16_t h[8]; uint4 q; };

__device__ __forceinline__ uint4 cvt8(const float* __restrict__ src) {
    float4 x = *(const float4*)src;
    float4 y = *(const float4*)(src + 4);
    PackBF u;
    u.h[0] = (bf16_t)x.x; u.h[1] = (bf16_t)x.y; u.h[2] = (bf16_t)x.z; u.h[3] = (bf16_t)x.w;
    u.h[4] = (bf16_t)y.x; u.h[5] = (bf16_t)y.y; u.h[6] = (bf16_t)y.z; u.h[7] = (bf16_t)y.w;
    return u.q;
}

// ---------------------------------------------------------------------------
// K0a: build bf16 concatenated activation matrices (8 cols / thread, 16B stores).
//   C1 = [X | b0 | a1]  (B x 3072)   C3 = [X | b1] (B x 2048)
//   C4 = [X | a1 | . ]  (B x 3072)   C5 = [X | . ] (B x 2048)
// ---------------------------------------------------------------------------
__global__ void pack_acts(const float* __restrict__ X,
                          const float* __restrict__ prev_a,
                          const float* __restrict__ prev_b,
                          bf16_t* __restrict__ C1, bf16_t* __restrict__ C3,
                          bf16_t* __restrict__ C4, bf16_t* __restrict__ C5) {
    int idx = blockIdx.x * 256 + threadIdx.x;   // NB*1024 threads, 8 cols each
    int b  = idx >> 10;
    int c  = (idx & 1023) * 8;                  // 0..8184, never straddles a source
    const float* src; bf16_t* dst;
    if (c < 3072) {
        src = (c < 1024) ? X + (size_t)b * 1024 + c
            : (c < 2048) ? prev_b + (size_t)b * 2048 + (c - 1024)            // b0
                         : prev_a + (size_t)b * 2048 + 1024 + (c - 2048);    // a1
        dst = C1 + (size_t)b * 3072 + c;
    } else if (c < 5120) {
        int cc = c - 3072;
        src = (cc < 1024) ? X + (size_t)b * 1024 + cc
                          : prev_b + (size_t)b * 2048 + 1024 + (cc - 1024);  // b1
        dst = C3 + (size_t)b * 2048 + cc;
    } else if (c < 7168) {
        int cc = c - 5120;
        src = (cc < 1024) ? X + (size_t)b * 1024 + cc
                          : prev_a + (size_t)b * 2048 + 1024 + (cc - 1024);  // a1
        dst = C4 + (size_t)b * 3072 + cc;
    } else {
        int cc = c - 7168;
        src = X + (size_t)b * 1024 + cc;
        dst = C5 + (size_t)b * 2048 + cc;
    }
    *(uint4*)dst = cvt8(src);
}

// K0b: fp32 -> bf16 weight conversion, 8 elems / thread
__global__ void cvt_bf16(const float* __restrict__ s, bf16_t* __restrict__ d, int n8) {
    int i = blockIdx.x * 256 + threadIdx.x;
    if (i < n8) *((uint4*)d + i) = cvt8(s + (size_t)i * 8);
}

// ---------------------------------------------------------------------------
// K1: C[M x N] = A[M x K](bf16,row-major,lda) * W[N x K](bf16,row-major,ldw)^T
//     fp32 accumulate via v_wmma_f32_16x16x32_bf16.
// Block: 256 threads (8 waves). Block tile 128(M) x 64(N); wave = 16M x 64N.
// Double-buffered LDS; global->LDS via async copies (ASYNCcnt) so the next
// K-slab streams in while the current one feeds 4 back-to-back WMMAs.
// ---------------------------------------------------------------------------
#define SBUF 15360           // one LDS buffer: 128*80 (A) + 64*80 (W)
__global__ __launch_bounds__(256) void gemm_bf16_wmma(
    const bf16_t* __restrict__ A, int lda,
    const bf16_t* __restrict__ W, int ldw,
    float* __restrict__ C, int N, int K) {
    __shared__ __align__(16) unsigned char smem[2 * SBUF];

    const int tid  = threadIdx.x;
    const int wave = tid >> 5;
    const int lane = tid & 31;
    const int half = lane >> 4;
    const int l16  = lane & 15;
    const int row0 = blockIdx.y * 128;
    const int col0 = blockIdx.x * 64;

    // Per-thread fill assignments: A tile 128 rows x 32 bf16 (2 chunks/thread),
    // W tile 64 rows x 32 bf16 (1 chunk/thread); 16B chunks, padded 80B rows.
    const int fr = tid >> 2, fc = tid & 3;
    const bf16_t* gA0 = A + (size_t)(row0 + fr) * lda + fc * 8;
    const bf16_t* gA1 = gA0 + (size_t)64 * lda;
    const bf16_t* gW  = W + (size_t)(col0 + fr) * ldw + fc * 8;
    const unsigned sbase = (unsigned)(uintptr_t)(const void*)smem;  // LDS byte addr
    const unsigned offA0 = fr * 80 + fc * 16;
    const unsigned offA1 = offA0 + 64 * 80;
    const unsigned offW  = 10240 + offA0;

    v8f acc[4] = {};

    const int T = K >> 5;   // K/32 slabs
    // prologue: stage slab 0 into buffer 0
    async_ld16(sbase + offA0, gA0);
    async_ld16(sbase + offA1, gA1);
    async_ld16(sbase + offW,  gW);

    for (int t = 0; t < T; ++t) {
        const int cur = t & 1;
        if (t + 1 < T) {
            // stream next slab into the other buffer while we compute
            const unsigned bo = (cur ^ 1) * SBUF;
            const size_t ke = (size_t)(t + 1) * 32;
            async_ld16(sbase + bo + offA0, gA0 + ke);
            async_ld16(sbase + bo + offA1, gA1 + ke);
            async_ld16(sbase + bo + offW,  gW  + ke);
            asm volatile("s_wait_asynccnt 0x3" ::: "memory");  // slab t resident
        } else {
            asm volatile("s_wait_asynccnt 0x0" ::: "memory");
        }
        __syncthreads();

        const unsigned char* bufp = smem + cur * SBUF;
        // Gather ALL fragments first, then issue 4 WMMAs back-to-back.
        union Frag { uint4 q[2]; v16bf v; } fa, fb0, fb1, fb2, fb3;
        // A fragment (16x32 bf16): lanes 0-15 hold K=0..15, lanes 16-31 K=16..31
        const unsigned char* ra = bufp + (wave * 16 + l16) * 80 + half * 32;
        fa.q[0] = *(const uint4*)(ra);
        fa.q[1] = *(const uint4*)(ra + 16);
        // B fragments (32x16): lanes 0-15 hold K=0..15, lanes 16-31 K=16..31
        const unsigned char* rb = bufp + 10240 + l16 * 80 + half * 32;
        fb0.q[0] = *(const uint4*)(rb);
        fb0.q[1] = *(const uint4*)(rb + 16);
        fb1.q[0] = *(const uint4*)(rb + 16 * 80);
        fb1.q[1] = *(const uint4*)(rb + 16 * 80 + 16);
        fb2.q[0] = *(const uint4*)(rb + 32 * 80);
        fb2.q[1] = *(const uint4*)(rb + 32 * 80 + 16);
        fb3.q[0] = *(const uint4*)(rb + 48 * 80);
        fb3.q[1] = *(const uint4*)(rb + 48 * 80 + 16);

        acc[0] = __builtin_amdgcn_wmma_f32_16x16x32_bf16(
            false, fa.v, false, fb0.v, (short)0, acc[0], false, false);
        acc[1] = __builtin_amdgcn_wmma_f32_16x16x32_bf16(
            false, fa.v, false, fb1.v, (short)0, acc[1], false, false);
        acc[2] = __builtin_amdgcn_wmma_f32_16x16x32_bf16(
            false, fa.v, false, fb2.v, (short)0, acc[2], false, false);
        acc[3] = __builtin_amdgcn_wmma_f32_16x16x32_bf16(
            false, fa.v, false, fb3.v, (short)0, acc[3], false, false);

        __syncthreads();   // done reading buf[cur]; safe to refill at t+2
    }

    // epilogue: C/D layout lane l16 = N, vgpr v -> M = v + 8*half
    #pragma unroll
    for (int n4 = 0; n4 < 4; ++n4) {
        #pragma unroll
        for (int v = 0; v < 8; ++v) {
            C[(size_t)(row0 + wave * 16 + v + 8 * half) * N + col0 + n4 * 16 + l16] =
                acc[n4][v];
        }
    }
}

// ---------------------------------------------------------------------------
// K2: gate phase-1 results into bf16 tails of phase-2 A matrices (8/thread).
// ---------------------------------------------------------------------------
__global__ void gate_pack(const float* __restrict__ ga00, const float* __restrict__ ga10,
                          const float* __restrict__ prev_a, const int* __restrict__ prev_depth,
                          bf16_t* __restrict__ C4, bf16_t* __restrict__ C5) {
    int idx = blockIdx.x * 256 + threadIdx.x;  // NB*128 threads, 8 h each
    int b = idx >> 7, h = (idx & 127) * 8;
    float pd = (float)prev_depth[b];
    size_t gi = (size_t)b * 1024 + h;
    PackBF u4, u5;
    #pragma unroll
    for (int i = 0; i < 8; ++i) {
        float a1 = prev_a[(size_t)b * 2048 + 1024 + h + i];
        u4.h[i] = (bf16_t)gatef(pd, 1.0f, ga00[gi + i], a1);
        u5.h[i] = (bf16_t)gatef(pd, 0.0f, ga10[gi + i], a1);
    }
    *(uint4*)(C4 + (size_t)b * 3072 + 2048 + h) = u4.q;
    *(uint4*)(C5 + (size_t)b * 2048 + 1024 + h) = u5.q;
}

// ---------------------------------------------------------------------------
// K3: per-row attention: logits (only nonzero 4x1024 slices of att_in),
// sigmoid -> softmax -> mask -> argmax -> sel / next_depth / f / j.
// One wave32 per row.
// ---------------------------------------------------------------------------
__global__ __launch_bounds__(256) void attn_select(
    const float* __restrict__ ga10, const float* __restrict__ gb10,
    const float* __restrict__ gb11,
    const float* __restrict__ prev_a, const float* __restrict__ prev_b,
    const int* __restrict__ prev_depth, const float* __restrict__ watt,
    float* __restrict__ sel_out,
    float* __restrict__ out_depth, float* __restrict__ out_f, float* __restrict__ out_j) {
    int wave = threadIdx.x >> 5, lane = threadIdx.x & 31;
    int row = blockIdx.x * 8 + wave;
    float pd = (float)prev_depth[row];
    float l[4] = {0.f, 0.f, 0.f, 0.f};
    for (int h = lane; h < 1024; h += 32) {
        size_t i = (size_t)row * 1024 + h;
        float a1 = prev_a[(size_t)row * 2048 + 1024 + h];
        float b1 = prev_b[(size_t)row * 2048 + 1024 + h];
        float na10 = gatef(pd, 0.0f, ga10[i], a1);
        float nb10 = gatef(pd, 0.0f, gb10[i], b1);
        float na11 = gatef(pd, 1.0f, a1, a1);
        float nb11 = gatef(pd, 1.0f, gb11[i], b1);
        #pragma unroll
        for (int k = 0; k < 4; ++k) {
            const float* wk = watt + (size_t)k * 8192;
            l[k] += wk[2048 + h] * na10 + wk[3072 + h] * nb10
                  + wk[6144 + h] * na11 + wk[7168 + h] * nb11;
        }
    }
    #pragma unroll
    for (int k = 0; k < 4; ++k)
        #pragma unroll
        for (int off = 16; off > 0; off >>= 1)
            l[k] += __shfl_xor(l[k], off, 32);

    if (lane == 0) {
        float s[4], e[4], av[4], mask[4];
        float mx = -1e30f;
        #pragma unroll
        for (int k = 0; k < 4; ++k) {
            s[k] = 1.0f / (1.0f + __expf(-l[k]));
            mx = fmaxf(mx, s[k]);
        }
        float sum = 0.f;
        #pragma unroll
        for (int k = 0; k < 4; ++k) { e[k] = __expf(s[k] - mx); sum += e[k]; }
        float e0 = (pd == 0.0f) ? 1.f : 0.f;
        float e1 = (pd == 1.0f) ? 1.f : 0.f;
        float eD = (pd == 1.0f) ? 1.f : 0.f;   // DEPTH == 1
        mask[0] = 1.f - e0; mask[1] = (1.f - e0) * (1.f - e1);
        mask[2] = 1.f - eD; mask[3] = 1.f - e0;
        int am = 0; float bv = -1e30f;
        #pragma unroll
        for (int k = 0; k < 4; ++k) {
            av[k] = mask[k] * e[k] / sum;
            if (av[k] > bv) { bv = av[k]; am = k; }  // first-max, like jnp.argmax
        }
        float sel[4];
        #pragma unroll
        for (int k = 0; k < 4; ++k) {
            float hard = (k == am) ? 1.0f : 0.0f;
            sel[k] = (hard - av[k]) + av[k];
            sel_out[(size_t)row * 4 + k] = sel[k];
        }
        int pdi = prev_depth[row];
        int nd = (am == 1) ? pdi - 1 : (am == 2) ? pdi + 1 : pdi;
        out_depth[row] = (float)nd;
        out_f[row] = sel[2] + sel[3];
        out_j[row] = sel[1] + sel[3];
    }
}

// ---------------------------------------------------------------------------
// K4: elementwise gates + sel-weighted combine -> next_a / next_b (float4).
// hidden[:, 0:2048] is identically zero (all state rows start with 2H zeros).
// ---------------------------------------------------------------------------
__global__ void combine(const float* __restrict__ ga00, const float* __restrict__ ga10,
                        const float* __restrict__ gb00, const float* __restrict__ gb11,
                        const float* __restrict__ gb10, const float* __restrict__ gb01,
                        const float* __restrict__ prev_a, const float* __restrict__ prev_b,
                        const int* __restrict__ prev_depth, const float* __restrict__ sel,
                        float* __restrict__ out_a, float* __restrict__ out_b) {
    int idx = blockIdx.x * 256 + threadIdx.x;  // NB*256 threads, 4 h each
    int b = idx >> 8, h = (idx & 255) * 4;
    float pd = (float)prev_depth[b];
    float s0 = sel[(size_t)b * 4], s1 = sel[(size_t)b * 4 + 1];
    float s2 = sel[(size_t)b * 4 + 2], s3 = sel[(size_t)b * 4 + 3];
    size_t gi = (size_t)b * 1024 + h;
    float4 va0 = *(const float4*)(prev_a + (size_t)b * 2048 + h);
    float4 va1 = *(const float4*)(prev_a + (size_t)b * 2048 + 1024 + h);
    float4 vb1 = *(const float4*)(prev_b + (size_t)b * 2048 + 1024 + h);
    float4 va00 = *(const float4*)(ga00 + gi);
    float4 va10 = *(const float4*)(ga10 + gi);
    float4 vb00 = *(const float4*)(gb00 + gi);
    float4 vb11 = *(const float4*)(gb11 + gi);
    float4 vb10 = *(const float4*)(gb10 + gi);
    float4 vb01 = *(const float4*)(gb01 + gi);
    float4 rA, rB;
    const float* pa0 = &va0.x;  const float* pa1 = &va1.x;  const float* pb1 = &vb1.x;
    const float* p00 = &va00.x; const float* p10 = &va10.x;
    const float* q00 = &vb00.x; const float* q11 = &vb11.x;
    const float* q10 = &vb10.x; const float* q01 = &vb01.x;
    float* oA = &rA.x; float* oB = &rB.x;
    #pragma unroll
    for (int i = 0; i < 4; ++i) {
        float a0 = pa0[i], a1 = pa1[i], b1v = pb1[i];
        float na00 = gatef(pd, 1.0f, p00[i], a1);
        float na11 = gatef(pd, 1.0f, a1,     a1);
        float na10 = gatef(pd, 0.0f, p10[i], a1);
        float na01 = gatef(pd, 2.0f, a0,     a1);
        float nb00 = gatef(pd, 1.0f, q00[i], b1v);
        float nb11 = gatef(pd, 1.0f, q11[i], b1v);
        float nb10 = gatef(pd, 0.0f, q10[i], b1v);
        float nb01 = gatef(pd, 2.0f, q01[i], b1v);
        oA[i] = s0 * na00 + s1 * na01 + s2 * na10 + s3 * na11;
        oB[i] = s0 * nb00 + s1 * nb01 + s2 * nb10 + s3 * nb11;
    }
    float4 z = make_float4(0.f, 0.f, 0.f, 0.f);
    *(float4*)(out_a + (size_t)b * 2048 + h) = z;
    *(float4*)(out_a + (size_t)b * 2048 + 1024 + h) = rA;
    *(float4*)(out_b + (size_t)b * 2048 + h) = z;
    *(float4*)(out_b + (size_t)b * 2048 + 1024 + h) = rB;
}

extern "C" void kernel_launch(void* const* d_in, const int* in_sizes, int n_in,
                              void* d_out, int out_size, void* d_ws, size_t ws_size,
                              hipStream_t stream) {
    const float* X          = (const float*)d_in[0];
    const float* prev_a     = (const float*)d_in[3];
    const float* prev_b     = (const float*)d_in[4];
    const int*   prev_depth = (const int*)d_in[5];
    const float* w_a00 = (const float*)d_in[6];
    const float* w_a10 = (const float*)d_in[7];
    const float* w_b00 = (const float*)d_in[8];
    const float* w_b11 = (const float*)d_in[9];
    const float* w_b10 = (const float*)d_in[10];
    const float* w_b01 = (const float*)d_in[11];
    const float* w_att = (const float*)d_in[12];

    unsigned char* ws = (unsigned char*)d_ws;
    bf16_t* C1   = (bf16_t*)(ws + 0);           // 4096*3072*2 = 25165824
    bf16_t* C3   = (bf16_t*)(ws + 25165824);    // 4096*2048*2 = 16777216
    bf16_t* C4   = (bf16_t*)(ws + 41943040);    // 25165824
    bf16_t* C5   = (bf16_t*)(ws + 67108864);    // 16777216
    bf16_t* Wa00 = (bf16_t*)(ws + 83886080);    // 6291456
    bf16_t* Wb01 = (bf16_t*)(ws + 90177536);    // 6291456
    bf16_t* Wb00 = (bf16_t*)(ws + 96468992);    // 6291456
    bf16_t* Wa10 = (bf16_t*)(ws + 102760448);   // 4194304
    bf16_t* Wb11 = (bf16_t*)(ws + 106954752);   // 4194304
    bf16_t* Wb10 = (bf16_t*)(ws + 111149056);   // 4194304
    float*  ga00 = (float*)(ws + 115343360);    // 16777216 each
    float*  ga10 = (float*)(ws + 132120576);
    float*  gb11 = (float*)(ws + 148897792);
    float*  gb01 = (float*)(ws + 165675008);
    float*  gb00 = (float*)(ws + 182452224);
    float*  gb10 = (float*)(ws + 199229440);
    float*  selb = (float*)(ws + 216006656);    // 4096*4*4

    float* out   = (float*)d_out;
    float* out_a = out;                 // 4096*2*1024
    float* out_b = out + 8388608;
    float* out_d = out + 16777216;
    float* out_f = out + 16781312;
    float* out_j = out + 16785408;

    // Phase 0: pack activations + convert weights to bf16 (16B vector stores)
    pack_acts<<<(NB * 1024) / 256, 256, 0, stream>>>(X, prev_a, prev_b, C1, C3, C4, C5);
    cvt_bf16<<<(393216 + 255) / 256, 256, 0, stream>>>(w_a00, Wa00, 393216);   // 3145728/8
    cvt_bf16<<<(393216 + 255) / 256, 256, 0, stream>>>(w_b01, Wb01, 393216);
    cvt_bf16<<<(393216 + 255) / 256, 256, 0, stream>>>(w_b00, Wb00, 393216);
    cvt_bf16<<<(262144 + 255) / 256, 256, 0, stream>>>(w_a10, Wa10, 262144);   // 2097152/8
    cvt_bf16<<<(262144 + 255) / 256, 256, 0, stream>>>(w_b11, Wb11, 262144);
    cvt_bf16<<<(262144 + 255) / 256, 256, 0, stream>>>(w_b10, Wb10, 262144);

    dim3 ggrid(NH / 64, NB / 128);  // (16, 32)
    // Phase 1 GEMMs
    gemm_bf16_wmma<<<ggrid, 256, 0, stream>>>(C1, 3072, Wa00, 3072, ga00, NH, 3072);
    gemm_bf16_wmma<<<ggrid, 256, 0, stream>>>(C1, 3072, Wb01, 3072, gb01, NH, 3072);
    gemm_bf16_wmma<<<ggrid, 256, 0, stream>>>(C1, 3072, Wa10, 2048, ga10, NH, 2048); // K=2048 prefix of C1
    gemm_bf16_wmma<<<ggrid, 256, 0, stream>>>(C3, 2048, Wb11, 2048, gb11, NH, 2048);

    // Phase 1.5: gate na00/na10 into phase-2 inputs
    gate_pack<<<(NB * 128) / 256, 256, 0, stream>>>(ga00, ga10, prev_a, prev_depth, C4, C5);

    // Phase 2 GEMMs
    gemm_bf16_wmma<<<ggrid, 256, 0, stream>>>(C4, 3072, Wb00, 3072, gb00, NH, 3072);
    gemm_bf16_wmma<<<ggrid, 256, 0, stream>>>(C5, 2048, Wb10, 2048, gb10, NH, 2048);

    // Phase 3: attention select + outputs
    attn_select<<<NB / 8, 256, 0, stream>>>(ga10, gb10, gb11, prev_a, prev_b, prev_depth,
                                            w_att, selb, out_d, out_f, out_j);
    combine<<<(NB * 256) / 256, 256, 0, stream>>>(ga00, ga10, gb00, gb11, gb10, gb01,
                                                  prev_a, prev_b, prev_depth, selb,
                                                  out_a, out_b);
}
